// LSTMScratch_31086973288867
// MI455X (gfx1250) — matrix operand order repaired and checked by reading, not compiled, not run on hardware
//
#include <hip/hip_runtime.h>

// ---------------------------------------------------------------------------
// LSTM on MI455X (gfx1250, wave32, WMMA).
//
// Phase 1: x_proj[T*B,4H] = X[T*B,D] @ Wx[D,4H] + b   -- huge parallel GEMM,
//          v_wmma_f32_16x16x32_f16 (f16 operands, f32 accum), X pre-converted
//          to f16 so the GEMM loop is pure b128 loads + wmma.
// Phase 2: persistent-kernel scan, 512 steps, device-wide spin barrier.
//          Block = 4 waves sharing one j-tile (16 h-columns); each wave owns
//          one m-tile and computes all 4 gate tiles. The block's 128 KB slice
//          of Wh (B-fragments) is staged into LDS ONCE and read via
//          ds_load_b128 on every step -> shortest possible per-step latency.
//          Cell state c lives in VGPRs for all 512 steps; h double-buffered.
//
// Workspace layout (~336.6 MB):
//   [0,256)          barrier counter (memset 0 each launch)
//   Wx_sw : 8   MB   Wx pre-swizzled to B-fragment order (f16)
//   Wh_sw : 8   MB   Wh pre-swizzled to B-fragment order (f16)
//   XP    : 256 MB   x_proj, f16, C-fragment tile order
//   X16   : 64  MB   inputs converted to f16, row-major
//   h16   : 256 KB   hidden state, 2 ping-pong buffers of [64,1024] f16
// ---------------------------------------------------------------------------

typedef _Float16 v16h __attribute__((ext_vector_type(16)));
typedef _Float16 v8h  __attribute__((ext_vector_type(8)));
typedef float    v8f  __attribute__((ext_vector_type(8)));
typedef float    v4f  __attribute__((ext_vector_type(4)));

#define T_STEPS 512
#define BATCH   64
#define DIM     1024      // D == H
#define NTILES  256       // 4H/16
#define KTILES  32        // DIM/32
#define SCAN_BLOCKS 64    // one block per j-tile
#define WH_LDS_BYTES (4 * KTILES * 512 * 2)   // 4 gates * 32 kt * 1KB = 128 KB

static __device__ inline v8f wmma_f16(v16h a, v16h b, v8f c) {
  return __builtin_amdgcn_wmma_f32_16x16x32_f16(
      /*neg_a=*/false, a, /*neg_b=*/false, b,
      /*c_mod=*/(short)0, c, /*reuse_a=*/false, /*reuse_b=*/false);
}

// A-fragment (16x32, MxK) from row-major f16. ISA layout: lane L holds row
// (L&15); k-octets {8*(L>>4)..+7} and {16+8*(L>>4)..+7} -> two 16B loads.
static __device__ inline v16h load_a_f16(const _Float16* __restrict__ A,
                                         int ldk, int row_base, int kt, int lane) {
  int m  = row_base + (lane & 15);
  int kb = kt * 32 + ((lane >> 4) << 3);
  const _Float16* p = A + (size_t)m * ldk + kb;
  v8h lo = *(const v8h*)p;
  v8h hi = *(const v8h*)(p + 16);
  v16h a;
#pragma unroll
  for (int i = 0; i < 8; ++i) { a[i] = lo[i]; a[8 + i] = hi[i]; }
  return a;
}

// B-fragment from pre-swizzled global weights: 32B contiguous per lane.
static __device__ inline v16h load_b(const _Float16* __restrict__ Wsw,
                                     int kt, int nt, int lane) {
  return *(const v16h*)(Wsw + ((((size_t)kt * NTILES + nt) * 32 + lane) << 4));
}

// B-fragment from LDS-staged weights (scan): tile index = g*32+kt.
static __device__ inline v16h load_b_lds(const _Float16* L, int g, int kt, int lane) {
  return *(const v16h*)(L + ((size_t)(g * KTILES + kt) * 512) + lane * 16);
}

static __device__ inline float fast_sigmoid(float x) {
  return 1.0f / (1.0f + __expf(-x));
}
static __device__ inline float fast_tanh(float x) {
  float e = __expf(-2.0f * x);
  return (1.0f - e) / (1.0f + e);
}

// ---------------------------------------------------------------------------
// Prep kernels
// ---------------------------------------------------------------------------

// Swizzle W (4 gates, each [1024,1024] f32 row-major, K-major) into
// B-fragment order f16: lane L => column n=(L&15), k = kt*32 + 8*(L>>4) + (i&7) + 16*(i>>3).
__global__ __launch_bounds__(256) void lstm_swizzle_w(
    const float* __restrict__ Wi, const float* __restrict__ Wf,
    const float* __restrict__ Wo, const float* __restrict__ Wc,
    _Float16* __restrict__ Wsw)
{
  int idx  = blockIdx.x * 256 + threadIdx.x;   // 0 .. 32*256*32-1
  int lane = idx & 31;
  int nt   = (idx >> 5) & (NTILES - 1);
  int kt   = idx >> 13;
  int gate = nt >> 6;
  int ncol = ((nt & 63) << 4) + (lane & 15);
  const float* W = (gate == 0) ? Wi : (gate == 1) ? Wf : (gate == 2) ? Wo : Wc;
  _Float16* dst = Wsw + (((size_t)kt * NTILES + nt) * 32 + lane) * 16;
  int k0 = kt * 32 + ((lane >> 4) << 3);
#pragma unroll
  for (int i = 0; i < 16; ++i)
    dst[i] = (_Float16)W[(size_t)(k0 + (i & 7) + ((i >> 3) << 4)) * DIM + ncol];
}

// Bulk f32 -> f16 convert of the inputs (8 elements / thread, vectorized).
__global__ __launch_bounds__(256) void lstm_cvt_x(
    const float* __restrict__ X, _Float16* __restrict__ X16)
{
  size_t i = ((size_t)blockIdx.x * 256 + threadIdx.x) * 8;
  v4f a = *(const v4f*)(X + i);
  v4f b = *(const v4f*)(X + i + 4);
  v8h o;
#pragma unroll
  for (int j = 0; j < 4; ++j) { o[j] = (_Float16)a[j]; o[4 + j] = (_Float16)b[j]; }
  *(v8h*)(X16 + i) = o;
}

__global__ __launch_bounds__(256) void lstm_init_h(
    const float* __restrict__ H0, _Float16* __restrict__ h16)
{
  int i = blockIdx.x * 256 + threadIdx.x;
  if (i < BATCH * DIM) h16[i] = (_Float16)H0[i];
}

// ---------------------------------------------------------------------------
// Phase 1: x_proj GEMM. One wave computes a 16x64 strip (1 mtile x 4 ntiles).
// 2048 rowtiles x 64 nblocks = 131072 waves = 16384 blocks of 256 threads.
// Output f16 in C-fragment tile order: tile (r,nt) at (r*256+nt)*256 halves,
// lane stores its 8 values contiguously at +lane*8.
// ---------------------------------------------------------------------------
__global__ __launch_bounds__(256) void lstm_xproj(
    const _Float16* __restrict__ X16, const _Float16* __restrict__ Wx_sw,
    const float* __restrict__ b_i, const float* __restrict__ b_f,
    const float* __restrict__ b_o, const float* __restrict__ b_c,
    _Float16* __restrict__ XP)
{
  int w    = blockIdx.x * 8 + (threadIdx.x >> 5);   // 0..131071
  int lane = threadIdx.x & 31;
  int r    = w >> 6;                                // rowtile 0..2047
  int nb   = w & 63;                                // 4-tile column block
  v8f acc0 = {}, acc1 = {}, acc2 = {}, acc3 = {};
  for (int kt = 0; kt < KTILES; ++kt) {
    v16h a = load_a_f16(X16, DIM, r * 16, kt, lane);
    acc0 = wmma_f16(a, load_b(Wx_sw, kt, nb * 4 + 0, lane), acc0);
    acc1 = wmma_f16(a, load_b(Wx_sw, kt, nb * 4 + 1, lane), acc1);
    acc2 = wmma_f16(a, load_b(Wx_sw, kt, nb * 4 + 2, lane), acc2);
    acc3 = wmma_f16(a, load_b(Wx_sw, kt, nb * 4 + 3, lane), acc3);
  }
  int lq = lane & 15;
  v8f* accs[4] = { &acc0, &acc1, &acc2, &acc3 };
#pragma unroll
  for (int q = 0; q < 4; ++q) {
    int nt   = nb * 4 + q;
    int gate = nt >> 6;
    int bidx = ((nt & 63) << 4) + lq;
    const float* bp = (gate == 0) ? b_i : (gate == 1) ? b_f : (gate == 2) ? b_o : b_c;
    float bv = bp[bidx];
    v8h o;
#pragma unroll
    for (int j = 0; j < 8; ++j) o[j] = (_Float16)((*accs[q])[j] + bv);
    *(v8h*)(XP + ((size_t)r * NTILES + nt) * 256 + lane * 8) = o;
  }
}

// ---------------------------------------------------------------------------
// Phase 2: persistent scan. 64 blocks x 128 threads. Block b owns j-tile
// jt = b (16 h-columns); wave w (= tid>>5) owns m-tile w. The block's Wh
// slice (4 gate tiles x 32 ktiles = 128 KB) is staged to LDS once.
// ---------------------------------------------------------------------------
__global__ __launch_bounds__(128) void lstm_scan(
    const _Float16* __restrict__ Wh_sw, const _Float16* __restrict__ XP,
    const float* __restrict__ C0, _Float16* __restrict__ h16,
    float* __restrict__ out, int* __restrict__ bar)
{
  extern __shared__ _Float16 WhLds[];   // 65536 halves = 128 KB (dynamic)

  int jt    = blockIdx.x;               // 0..63 : j-tile (h columns)
  int mtile = threadIdx.x >> 5;         // 0..3  : m-tile (batch rows)
  int lane  = threadIdx.x & 31;
  int half  = lane >> 4, lq = lane & 15;
  int col   = jt * 16 + lq;             // h-feature column owned by this lane
  int rowb  = mtile * 16 + half * 8;

  // ---- one-time LDS stage of this block's Wh fragments ----
  for (int c = threadIdx.x; c < 8192; c += 128) {       // 8192 x v8h = 128 KB
    int tile   = c >> 6;                                // g*32 + kt
    int kt     = tile & (KTILES - 1);
    int g      = tile >> 5;
    int within = (c & 63) * 8;                          // halves inside tile
    const _Float16* src =
        Wh_sw + (((size_t)kt * NTILES + (g * 64 + jt)) * 32) * 16 + within;
    *(v8h*)(WhLds + (size_t)tile * 512 + within) = *(const v8h*)src;
  }
  __syncthreads();

  float cc[8], hv[8];
#pragma unroll
  for (int j = 0; j < 8; ++j)
    cc[j] = C0[(size_t)(rowb + j) * DIM + col];

  for (int t = 0; t < T_STEPS; ++t) {
    const _Float16* hrd = h16 + (size_t)(t & 1) * (BATCH * DIM);
    _Float16*       hwr = h16 + (size_t)((t + 1) & 1) * (BATCH * DIM);

    v8f aI = {}, aF = {}, aO = {}, aG = {};
#pragma unroll 2
    for (int kt = 0; kt < KTILES; ++kt) {
      v16h a = load_a_f16(hrd, DIM, mtile * 16, kt, lane);
      aI = wmma_f16(a, load_b_lds(WhLds, 0, kt, lane), aI);
      aF = wmma_f16(a, load_b_lds(WhLds, 1, kt, lane), aF);
      aO = wmma_f16(a, load_b_lds(WhLds, 2, kt, lane), aO);
      aG = wmma_f16(a, load_b_lds(WhLds, 3, kt, lane), aG);
    }

    size_t rt = (size_t)(t * 4 + mtile) * NTILES;
    v8h xI = *(const v8h*)(XP + (rt +       jt) * 256 + lane * 8);
    v8h xF = *(const v8h*)(XP + (rt +  64 + jt) * 256 + lane * 8);
    v8h xO = *(const v8h*)(XP + (rt + 128 + jt) * 256 + lane * 8);
    v8h xG = *(const v8h*)(XP + (rt + 192 + jt) * 256 + lane * 8);

#pragma unroll
    for (int j = 0; j < 8; ++j) {
      float gi = aI[j] + (float)xI[j];
      float gf = aF[j] + (float)xF[j];
      float go = aO[j] + (float)xO[j];
      float gg = aG[j] + (float)xG[j];
      float iv = fast_sigmoid(gi);
      float fv = fast_sigmoid(gf);
      float ov = fast_sigmoid(go);
      float gv = fast_tanh(gg);
      cc[j] = fv * cc[j] + iv * gv;
      hv[j] = ov * fast_tanh(cc[j]);
      int rowg = rowb + j;
      hwr[(size_t)rowg * DIM + col] = (_Float16)hv[j];
      out[(size_t)t * (BATCH * DIM) + (size_t)rowg * DIM + col] = hv[j];
    }

    // ---- device-wide barrier between recurrent steps ----
    __threadfence();            // release this wave's h stores to device scope
    __syncthreads();
    if (threadIdx.x == 0) {
      __hip_atomic_fetch_add(bar, 1, __ATOMIC_RELEASE, __HIP_MEMORY_SCOPE_AGENT);
      while (__hip_atomic_load(bar, __ATOMIC_ACQUIRE, __HIP_MEMORY_SCOPE_AGENT)
             < SCAN_BLOCKS * (t + 1))
        __builtin_amdgcn_s_sleep(1);
    }
    __syncthreads();
    __threadfence();            // acquire: don't read stale h through L0
  }

  // Final Hf, Cf appended after outputs in d_out.
  float* Hf = out + (size_t)T_STEPS * (BATCH * DIM);
  float* Cf = Hf + (BATCH * DIM);
#pragma unroll
  for (int j = 0; j < 8; ++j) {
    Hf[(size_t)(rowb + j) * DIM + col] = hv[j];
    Cf[(size_t)(rowb + j) * DIM + col] = cc[j];
  }
}

extern "C" void kernel_launch(void* const* d_in, const int* in_sizes, int n_in,
                              void* d_out, int out_size, void* d_ws, size_t ws_size,
                              hipStream_t stream) {
  (void)in_sizes; (void)n_in; (void)out_size; (void)ws_size;

  const float* X    = (const float*)d_in[0];
  const float* H0   = (const float*)d_in[1];
  const float* C0   = (const float*)d_in[2];
  const float* W_xi = (const float*)d_in[3];
  const float* W_hi = (const float*)d_in[4];
  const float* b_i  = (const float*)d_in[5];
  const float* W_xf = (const float*)d_in[6];
  const float* W_hf = (const float*)d_in[7];
  const float* b_f  = (const float*)d_in[8];
  const float* W_xo = (const float*)d_in[9];
  const float* W_ho = (const float*)d_in[10];
  const float* b_o  = (const float*)d_in[11];
  const float* W_xc = (const float*)d_in[12];
  const float* W_hc = (const float*)d_in[13];
  const float* b_c  = (const float*)d_in[14];

  char* ws = (char*)d_ws;
  int*      bar   = (int*)ws;
  _Float16* Wx_sw = (_Float16*)(ws + 256);
  _Float16* Wh_sw = Wx_sw + (size_t)KTILES * NTILES * 32 * 16;       // +4 Mi halves
  _Float16* XP    = Wh_sw + (size_t)KTILES * NTILES * 32 * 16;       // +4 Mi halves
  _Float16* X16   = XP    + (size_t)T_STEPS * 4 * NTILES * 256;      // +128 Mi halves
  _Float16* h16   = X16   + (size_t)T_STEPS * BATCH * DIM;           // +32 Mi halves

  hipMemsetAsync(bar, 0, 256, stream);
  lstm_swizzle_w<<<1024, 256, 0, stream>>>(W_xi, W_xf, W_xo, W_xc, Wx_sw);
  lstm_swizzle_w<<<1024, 256, 0, stream>>>(W_hi, W_hf, W_ho, W_hc, Wh_sw);
  lstm_cvt_x<<<16384, 256, 0, stream>>>(X, X16);
  lstm_init_h<<<256, 256, 0, stream>>>(H0, h16);
  lstm_xproj<<<16384, 256, 0, stream>>>(X16, Wx_sw, b_i, b_f, b_o, b_c, XP);
  lstm_scan<<<SCAN_BLOCKS, 128, WH_LDS_BYTES, stream>>>(Wh_sw, XP, C0, h16,
                                                        (float*)d_out, bar);
}
